// Trans_197568495816
// MI455X (gfx1250) — matrix operand order repaired and checked by reading, not compiled, not run on hardware
//
#include <hip/hip_runtime.h>
#include <hip/hip_bf16.h>

// ---------------- problem constants ----------------
#define Bc     2
#define Tc     2048
#define INDIM  2048
#define Hc     512
#define NHc    8
#define DHc    64
#define Lc     6
#define NCc    48
#define FFc    2048
#define TTOT   (NCc + Tc)      // 2096
#define MTOT   (Bc * TTOT)     // 4192
#define QKVW   (3 * Hc)        // 1536

// ---------------- vector types ----------------
typedef __attribute__((ext_vector_type(16))) __bf16 v16bf;
typedef __attribute__((ext_vector_type(8)))  __bf16 v8bf;
typedef __attribute__((ext_vector_type(8)))  float  v8f;
typedef __attribute__((ext_vector_type(4)))  float  v4f;

union V16 { v16bf v; v8bf h[2]; };

__device__ __forceinline__ v8f wmma_bf16(const v16bf& a, const v16bf& b, const v8f& c) {
    return __builtin_amdgcn_wmma_f32_16x16x32_bf16(false, a, false, b, (short)0, c, false, false);
}

__device__ __forceinline__ v8bf pack8(const v4f& a, const v4f& b) {
    v8bf r;
    r[0] = (__bf16)a.x; r[1] = (__bf16)a.y; r[2] = (__bf16)a.z; r[3] = (__bf16)a.w;
    r[4] = (__bf16)b.x; r[5] = (__bf16)b.y; r[6] = (__bf16)b.z; r[7] = (__bf16)b.w;
    return r;
}

// ---------------- DPP 16-lane butterfly reductions (no LDS) ----------------
template<int CTRL>
__device__ __forceinline__ float dppf(float v) {
    return __int_as_float(__builtin_amdgcn_update_dpp(
        0, __float_as_int(v), CTRL, 0xF, 0xF, true));
}
__device__ __forceinline__ float red_max16(float v) {
    v = fmaxf(v, dppf<0xB1>(v));   // quad_perm(1,0,3,2)  : xor 1
    v = fmaxf(v, dppf<0x4E>(v));   // quad_perm(2,3,0,1)  : xor 2
    v = fmaxf(v, dppf<0x141>(v));  // row_half_mirror     : xor 4 (post-fold)
    v = fmaxf(v, dppf<0x140>(v));  // row_mirror          : xor 8 (post-fold)
    return v;
}
__device__ __forceinline__ float red_sum16(float v) {
    v += dppf<0xB1>(v);
    v += dppf<0x4E>(v);
    v += dppf<0x141>(v);
    v += dppf<0x140>(v);
    return v;
}

// =====================================================================
// Generic GEMM:  Y[M,N] = act( X[M,K] * W[N,K]^T + bias ) [+ res]
// 64x64 tile, 8 waves, K-step 32, double-buffered LDS, b128 staging.
// =====================================================================
#define TKM 64
#define TKN 64
#define TKK 32
#define LSTR 48   // halves per LDS row (96B => 16B-aligned fragment chunks)

template<int ACT, int HASRES>
__global__ __launch_bounds__(256) void gemm_wmma(
    const float* __restrict__ X, const float* __restrict__ W,
    const float* __restrict__ bias, const float* __restrict__ res,
    float* __restrict__ Y, int M, int N, int K)
{
    __shared__ __attribute__((aligned(16))) __bf16 Xs[2][TKM * LSTR];
    __shared__ __attribute__((aligned(16))) __bf16 Ws[2][TKN * LSTR];

    const int tid  = threadIdx.x;
    const int lane = tid & 31;
    const int wave = tid >> 5;      // 0..7
    const int wm   = wave >> 1;     // 0..3  -> 16-row slice
    const int wn   = wave & 1;      // 0..1  -> 32-col slice
    const int m0   = blockIdx.y * TKM;
    const int n0   = blockIdx.x * TKN;

    const int r = tid >> 2;          // 0..63 : tile row
    const int c = (tid & 3) * 8;     // 0,8,16,24 : tile col (K)

    // clamp staging rows into range: unconditional b128 loads
    const int gmr = (m0 + r < M) ? (m0 + r) : (M - 1);
    const int gnr = (n0 + r < N) ? (n0 + r) : (N - 1);
    const float* xrow = X + (size_t)gmr * K + c;
    const float* wrow = W + (size_t)gnr * K + c;

    v4f x0, x1, w0, w1;

    // prefetch K-tile 0
    x0 = *(const v4f*)(xrow);
    x1 = *(const v4f*)(xrow + 4);
    w0 = *(const v4f*)(wrow);
    w1 = *(const v4f*)(wrow + 4);
    *(v8bf*)&Xs[0][r * LSTR + c] = pack8(x0, x1);
    *(v8bf*)&Ws[0][r * LSTR + c] = pack8(w0, w1);

    v8f acc0 = {}; v8f acc1 = {};

    const int arow = wm * 16 + (lane & 15);
    const int kb   = (lane < 16) ? 0 : 8;    // unused: A also reads 16 contiguous K at kb2
    const int kb2  = (lane < 16) ? 0 : 16;   // B: 16 contiguous K at kb2
    const int b0r  = wn * 32 + (lane & 15);

    const int nsteps = K / TKK;
    for (int s = 0; s < nsteps; ++s) {
        __syncthreads();
        const int cur  = s & 1;
        const bool more = (s + 1) < nsteps;

        if (more) {
            const int ko = (s + 1) * TKK;
            x0 = *(const v4f*)(xrow + ko);
            x1 = *(const v4f*)(xrow + ko + 4);
            w0 = *(const v4f*)(wrow + ko);
            w1 = *(const v4f*)(wrow + ko + 4);
        }

        V16 af, bf0, bf1;
        af.h[0]  = *(const v8bf*)&Xs[cur][arow * LSTR + kb2];
        af.h[1]  = *(const v8bf*)&Xs[cur][arow * LSTR + kb2 + 8];
        bf0.h[0] = *(const v8bf*)&Ws[cur][b0r * LSTR + kb2];
        bf0.h[1] = *(const v8bf*)&Ws[cur][b0r * LSTR + kb2 + 8];
        bf1.h[0] = *(const v8bf*)&Ws[cur][(b0r + 16) * LSTR + kb2];
        bf1.h[1] = *(const v8bf*)&Ws[cur][(b0r + 16) * LSTR + kb2 + 8];

        acc0 = wmma_bf16(af.v, bf0.v, acc0);
        acc1 = wmma_bf16(af.v, bf1.v, acc1);

        if (more) {
            *(v8bf*)&Xs[1 - cur][r * LSTR + c] = pack8(x0, x1);
            *(v8bf*)&Ws[1 - cur][r * LSTR + c] = pack8(w0, w1);
        }
    }

    // store: lane covers col n=(lane&15), rows (lane>>4)*8 + j
    const int nloc  = lane & 15;
    const int mbase = (lane >> 4) * 8;
    #pragma unroll
    for (int f = 0; f < 2; ++f) {
        const v8f a = f ? acc1 : acc0;
        const int gn = n0 + wn * 32 + f * 16 + nloc;
        if (gn >= N) continue;
        const float bv = bias[gn];
        #pragma unroll
        for (int j = 0; j < 8; ++j) {
            const int gm = m0 + wm * 16 + mbase + j;
            if (gm >= M) continue;
            float v = a[j] + bv;
            if (ACT) v = fmaxf(v, 0.f);
            if (HASRES) v += res[(size_t)gm * N + gn];
            Y[(size_t)gm * N + gn] = v;
        }
    }
}

// =====================================================================
// LayerNorm over H=512, one block per row.
// =====================================================================
__global__ __launch_bounds__(256) void layernorm_k(
    const float* __restrict__ x, const float* __restrict__ g,
    const float* __restrict__ b, float* __restrict__ y)
{
    __shared__ float2 red[256];
    const int row = blockIdx.x;
    const int t   = threadIdx.x;
    const float* xr = x + (size_t)row * Hc;
    const float a0 = xr[t], a1 = xr[t + 256];
    red[t] = make_float2(a0 + a1, a0 * a0 + a1 * a1);
    __syncthreads();
    for (int off = 128; off > 0; off >>= 1) {
        if (t < off) {
            float2 o = red[t + off]; float2 m = red[t];
            m.x += o.x; m.y += o.y; red[t] = m;
        }
        __syncthreads();
    }
    const float mean = red[0].x * (1.f / Hc);
    const float var  = red[0].y * (1.f / Hc) - mean * mean;
    const float rstd = rsqrtf(var + 1e-5f);
    y[(size_t)row * Hc + t]       = (a0 - mean) * rstd * g[t]       + b[t];
    y[(size_t)row * Hc + t + 256] = (a1 - mean) * rstd * g[t + 256] + b[t + 256];
}

// =====================================================================
// tokens[b, tok, :] = (tok < NC) ? cls_tok[tok] : xproj[b, tok-NC] + pos[tok-NC]
// =====================================================================
__global__ void assemble_tokens(const float* __restrict__ xproj,
                                const float* __restrict__ pos,
                                const float* __restrict__ cls_tok,
                                float* __restrict__ tokens)
{
    const size_t idx = (size_t)blockIdx.x * blockDim.x + threadIdx.x;
    if (idx >= (size_t)Bc * TTOT * Hc) return;
    const int hh  = (int)(idx % Hc);
    const size_t r = idx / Hc;
    const int tok = (int)(r % TTOT);
    const int b   = (int)(r / TTOT);
    float v;
    if (tok < NCc) v = cls_tok[tok * Hc + hh];
    else v = xproj[((size_t)b * Tc + tok - NCc) * Hc + hh]
           + pos[(size_t)(tok - NCc) * Hc + hh];
    tokens[idx] = v;
}

// =====================================================================
// Flash attention: one wave per 16-query tile per (b, head).
// Processes 32 keys (two 16-key fragments) per iteration:
//   S: 4 WMMAs; joint softmax (DPP reductions); P is a dense 16x32
//   A-fragment; P*V: 4 dense WMMAs with all 32 lanes loading V.
// =====================================================================
__global__ __launch_bounds__(256) void attn_wmma(
    const float* __restrict__ qkv, const unsigned char* __restrict__ mask,
    float* __restrict__ attn_out, int local_r)
{
    __shared__ __attribute__((aligned(16))) __bf16 Plds[8 * 512]; // 16x32 per wave

    const int lane = threadIdx.x & 31;
    const int wave = threadIdx.x >> 5;
    const int wt   = blockIdx.x * 8 + wave;          // 0 .. B*NH*131-1 (exact)
    const int qt   = wt % (TTOT / 16);
    const int bh   = wt / (TTOT / 16);
    const int hidx = bh % NHc;
    const int bidx = bh / NHc;
    const int q0   = qt * 16;

    const int l15  = lane & 15;
    const int kb   = (lane < 16) ? 0 : 8;    // unused; A fragments also use kb2
    const int kb2  = (lane < 16) ? 0 : 16;   // B-fragment K-range select

    // Q fragments (A layout) for the two dh chunks, vectorized loads
    V16 qa0, qa1;
    {
        const float* qp = qkv + ((size_t)bidx * TTOT + (q0 + l15)) * QKVW + hidx * DHc;
        const v4f a0 = *(const v4f*)(qp + kb2);
        const v4f a1 = *(const v4f*)(qp + kb2 + 4);
        const v4f a2 = *(const v4f*)(qp + kb2 + 8);
        const v4f a3 = *(const v4f*)(qp + kb2 + 12);
        const v4f a4 = *(const v4f*)(qp + 32 + kb2);
        const v4f a5 = *(const v4f*)(qp + 32 + kb2 + 4);
        const v4f a6 = *(const v4f*)(qp + 32 + kb2 + 8);
        const v4f a7 = *(const v4f*)(qp + 32 + kb2 + 12);
        qa0.h[0] = pack8(a0, a1); qa0.h[1] = pack8(a2, a3);
        qa1.h[0] = pack8(a4, a5); qa1.h[1] = pack8(a6, a7);
    }

    // per-row query validity (tok_mask), rows j held by this lane
    int tq[8];
    #pragma unroll
    for (int j = 0; j < 8; ++j) {
        const int qg = q0 + ((lane >> 4) * 8) + j;
        tq[j] = (qg < NCc) ? 1 : (int)mask[(size_t)bidx * Tc + (qg - NCc)];
    }
    const int myq = q0 + l15;
    const int any_pad =
        __any(((myq < NCc) ? 1 : (int)mask[(size_t)bidx * Tc + (myq - NCc)]) == 0);

    float mrun[8], lrun[8];
    v8f o[4] = {v8f{}, v8f{}, v8f{}, v8f{}};
    #pragma unroll
    for (int j = 0; j < 8; ++j) { mrun[j] = -1e30f; lrun[j] = 0.f; }

    const bool qcls = (q0 < NCc);
    const int NKT = (TTOT + 31) / 32;   // 66 pair-tiles

    for (int kt = 0; kt < NKT; ++kt) {
        const int k0 = kt * 32;
        bool need = qcls || (k0 < NCc) || (any_pad != 0);
        if (!need) {
            const int qlo = q0 - NCc, qhi = q0 + 15 - NCc;
            const int klo = k0 - NCc, khi = k0 + 31 - NCc;
            need = (klo <= qhi + local_r) && (khi >= qlo - local_r);
        }
        if (!need) continue;

        const bool full = (k0 + 32) <= TTOT;   // uniform

        // ---- S = Q * K^T for the two 16-key fragments ----
        v8f s0 = {}, s1 = {};
        {
            const int krow0 = k0 + l15;
            const int krow1c = (krow0 + 16 < TTOT) ? (krow0 + 16) : (TTOT - 1);
            const float* kp0 = qkv + ((size_t)bidx * TTOT + krow0)  * QKVW + Hc + hidx * DHc;
            const float* kp1 = qkv + ((size_t)bidx * TTOT + krow1c) * QKVW + Hc + hidx * DHc;
            V16 kf;
            kf.h[0] = pack8(*(const v4f*)(kp0 + kb2),     *(const v4f*)(kp0 + kb2 + 4));
            kf.h[1] = pack8(*(const v4f*)(kp0 + kb2 + 8), *(const v4f*)(kp0 + kb2 + 12));
            s0 = wmma_bf16(qa0.v, kf.v, s0);
            kf.h[0] = pack8(*(const v4f*)(kp0 + 32 + kb2),     *(const v4f*)(kp0 + 32 + kb2 + 4));
            kf.h[1] = pack8(*(const v4f*)(kp0 + 32 + kb2 + 8), *(const v4f*)(kp0 + 32 + kb2 + 12));
            s0 = wmma_bf16(qa1.v, kf.v, s0);
            kf.h[0] = pack8(*(const v4f*)(kp1 + kb2),     *(const v4f*)(kp1 + kb2 + 4));
            kf.h[1] = pack8(*(const v4f*)(kp1 + kb2 + 8), *(const v4f*)(kp1 + kb2 + 12));
            s1 = wmma_bf16(qa0.v, kf.v, s1);
            kf.h[0] = pack8(*(const v4f*)(kp1 + 32 + kb2),     *(const v4f*)(kp1 + 32 + kb2 + 4));
            kf.h[1] = pack8(*(const v4f*)(kp1 + 32 + kb2 + 8), *(const v4f*)(kp1 + 32 + kb2 + 12));
            s1 = wmma_bf16(qa1.v, kf.v, s1);
        }

        // ---- mask + joint online softmax over 32 keys ----
        const int kg0 = k0 + l15;
        const int kg1 = kg0 + 16;
        const int tk0 = (kg0 < NCc) ? 1 : (int)mask[(size_t)bidx * Tc + (kg0 - NCc)];
        const int tk1 = (kg1 < NCc) ? 1
                      : ((kg1 < TTOT) ? (int)mask[(size_t)bidx * Tc + (kg1 - NCc)] : 0);
        float pp0[8], pp1[8], alph[8];
        #pragma unroll
        for (int j = 0; j < 8; ++j) {
            const int m  = ((lane >> 4) * 8) + j;
            const int qg = q0 + m;
            const int qok = (!tq[j]) || (qg < NCc);
            const int ok0 = tk0 && (qok || (kg0 < NCc) ||
                                    (abs((qg - NCc) - (kg0 - NCc)) <= local_r));
            const int ok1 = tk1 && (qok || (kg1 < NCc) ||
                                    (abs((qg - NCc) - (kg1 - NCc)) <= local_r));
            const float sv0 = ok0 ? s0[j] * 0.125f : -1e30f;
            const float sv1 = ok1 ? s1[j] * 0.125f : -1e30f;
            const float rmax  = red_max16(fmaxf(sv0, sv1));
            const float newm  = fmaxf(mrun[j], rmax);
            const float alpha = __expf(mrun[j] - newm);
            const float pv0   = ok0 ? __expf(sv0 - newm) : 0.f;
            const float pv1   = ok1 ? __expf(sv1 - newm) : 0.f;
            const float psum  = red_sum16(pv0 + pv1);
            lrun[j] = lrun[j] * alpha + psum;
            mrun[j] = newm;
            alph[j] = alpha;
            pp0[j]  = pv0;
            pp1[j]  = pv1;
        }
        #pragma unroll
        for (int cc = 0; cc < 4; ++cc)
            #pragma unroll
            for (int j = 0; j < 8; ++j) o[cc][j] *= alph[j];

        // ---- transpose P (16x32) through LDS into A layout ----
        #pragma unroll
        for (int j = 0; j < 8; ++j) {
            const int m = ((lane >> 4) * 8) + j;
            Plds[wave * 512 + m * 32 + l15]      = (__bf16)pp0[j];
            Plds[wave * 512 + m * 32 + l15 + 16] = (__bf16)pp1[j];
        }
        asm volatile("s_wait_dscnt 0x0" ::: "memory");
        V16 pa;
        pa.h[0] = *(const v8bf*)&Plds[wave * 512 + l15 * 32 + kb2];
        pa.h[1] = *(const v8bf*)&Plds[wave * 512 + l15 * 32 + kb2 + 8];

        // ---- O += P * V (dense 32-K, 4 dh chunks of 16) ----
        const int vbase = ((bidx * TTOT + k0 + kb2) * QKVW) + 2 * Hc + hidx * DHc + l15;
        #pragma unroll
        for (int cc = 0; cc < 4; ++cc) {
            V16 vf;
            if (full) {
                #pragma unroll
                for (int u = 0; u < 16; ++u)
                    vf.v[u] = (__bf16)qkv[vbase + u * QKVW + cc * 16];
            } else {
                #pragma unroll
                for (int u = 0; u < 16; ++u) {
                    int vr = k0 + kb2 + u;
                    vr = (vr < TTOT) ? vr : (TTOT - 1);  // clamped rows have P=0
                    vf.v[u] = (__bf16)qkv[((size_t)bidx * TTOT + vr) * QKVW
                                          + 2 * Hc + hidx * DHc + cc * 16 + l15];
                }
            }
            o[cc] = wmma_bf16(pa.v, vf.v, o[cc]);
        }
    }

    // ---- normalize and write ----
    #pragma unroll
    for (int j = 0; j < 8; ++j) {
        const int m = ((lane >> 4) * 8) + j;
        const float linv = 1.f / fmaxf(lrun[j], 1e-20f);
        float* orow = attn_out + ((size_t)bidx * TTOT + q0 + m) * Hc + hidx * DHc;
        #pragma unroll
        for (int cc = 0; cc < 4; ++cc)
            orow[cc * 16 + l15] = o[cc][j] * linv;
    }
}

// =====================================================================
// pack outputs: [tok_cls (B*NC)] [cls_prob tokens (B*T*NC)] [tokens (B*TTOT*H)]
// =====================================================================
__global__ void assemble_out(const float* __restrict__ cls_prob,
                             const float* __restrict__ tokens,
                             float* __restrict__ out)
{
    const size_t n1 = (size_t)Bc * NCc;
    const size_t n2 = (size_t)Bc * Tc * NCc;
    const size_t n3 = (size_t)Bc * TTOT * Hc;
    const size_t idx = (size_t)blockIdx.x * blockDim.x + threadIdx.x;
    if (idx < n1) {
        const int b = (int)(idx / NCc), c = (int)(idx % NCc);
        out[idx] = cls_prob[((size_t)b * TTOT + c) * NCc + c];
    } else if (idx < n1 + n2) {
        const size_t j = idx - n1;
        const int c = (int)(j % NCc);
        const size_t r = j / NCc;
        const int t = (int)(r % Tc), b = (int)(r / Tc);
        out[idx] = cls_prob[((size_t)b * TTOT + NCc + t) * NCc + c];
    } else if (idx < n1 + n2 + n3) {
        out[idx] = tokens[idx - n1 - n2];
    }
}

// =====================================================================
static inline void launch_gemm(const float* X, const float* W, const float* bias,
                               const float* res, float* Y, int M, int N, int K,
                               int act, hipStream_t stream)
{
    const dim3 blk(256);
    const dim3 grid((N + TKN - 1) / TKN, (M + TKM - 1) / TKM);
    if (res)      gemm_wmma<0, 1><<<grid, blk, 0, stream>>>(X, W, bias, res, Y, M, N, K);
    else if (act) gemm_wmma<1, 0><<<grid, blk, 0, stream>>>(X, W, bias, nullptr, Y, M, N, K);
    else          gemm_wmma<0, 0><<<grid, blk, 0, stream>>>(X, W, bias, nullptr, Y, M, N, K);
}

extern "C" void kernel_launch(void* const* d_in, const int* in_sizes, int n_in,
                              void* d_out, int out_size, void* d_ws, size_t ws_size,
                              hipStream_t stream)
{
    (void)in_sizes; (void)n_in; (void)out_size; (void)ws_size;
    const float*        x       = (const float*)d_in[0];
    const unsigned char* maskp  = (const unsigned char*)d_in[1];
    const float* w_in  = (const float*)d_in[2];
    const float* b_in  = (const float*)d_in[3];
    const float* pos   = (const float*)d_in[4];
    const float* cls_t = (const float*)d_in[5];
    const float* w_cls = (const float*)d_in[6];
    const float* b_cls = (const float*)d_in[7];
    const float* ln1_g = (const float*)d_in[8];
    const float* ln1_b = (const float*)d_in[9];
    const float* ln2_g = (const float*)d_in[10];
    const float* ln2_b = (const float*)d_in[11];
    const float* w_qkv = (const float*)d_in[12];
    const float* b_qkv = (const float*)d_in[13];
    const float* w_o   = (const float*)d_in[14];
    const float* b_o   = (const float*)d_in[15];
    const float* w_ff1 = (const float*)d_in[16];
    const float* b_ff1 = (const float*)d_in[17];
    const float* w_ff2 = (const float*)d_in[18];
    const float* b_ff2 = (const float*)d_in[19];
    float* outp = (float*)d_out;

    float* ws      = (float*)d_ws;
    float* tokens  = ws;                                   // MTOT*H
    float* hbuf    = tokens + (size_t)MTOT * Hc;           // MTOT*H
    float* qkvb    = hbuf   + (size_t)MTOT * Hc;           // MTOT*3H
    float* attnb   = qkvb   + (size_t)MTOT * QKVW;         // MTOT*H
    float* ffb     = attnb  + (size_t)MTOT * Hc;           // MTOT*FF
    float* xproj   = ffb;                                  // alias (used pre-layers)
    float* clsb    = ffb    + (size_t)MTOT * FFc;          // MTOT*NC

    const dim3 blk(256);

    // 1) input projection: xproj[B*T, H] = x * w_in^T + b_in
    launch_gemm(x, w_in, b_in, nullptr, xproj, Bc * Tc, Hc, INDIM, 0, stream);

    // 2) tokens = [cls; xproj + pos]
    assemble_tokens<<<dim3(((size_t)Bc * TTOT * Hc + 255) / 256), blk, 0, stream>>>(
        xproj, pos, cls_t, tokens);

    for (int i = 0; i < Lc; ++i) {
        int local_r = 1 << i; if (local_r > Tc - 1) local_r = Tc - 1;

        layernorm_k<<<dim3(MTOT), blk, 0, stream>>>(
            tokens, ln1_g + i * Hc, ln1_b + i * Hc, hbuf);

        launch_gemm(hbuf, w_qkv + (size_t)i * QKVW * Hc, b_qkv + (size_t)i * QKVW,
                    nullptr, qkvb, MTOT, QKVW, Hc, 0, stream);

        attn_wmma<<<dim3((Bc * NHc * (TTOT / 16)) / 8), blk, 0, stream>>>(
            qkvb, maskp, attnb, local_r);

        launch_gemm(attnb, w_o + (size_t)i * Hc * Hc, b_o + (size_t)i * Hc,
                    tokens, tokens, MTOT, Hc, Hc, 0, stream);

        layernorm_k<<<dim3(MTOT), blk, 0, stream>>>(
            tokens, ln2_g + i * Hc, ln2_b + i * Hc, hbuf);

        launch_gemm(hbuf, w_ff1 + (size_t)i * FFc * Hc, b_ff1 + (size_t)i * FFc,
                    nullptr, ffb, MTOT, FFc, Hc, 1, stream);

        launch_gemm(ffb, w_ff2 + (size_t)i * Hc * FFc, b_ff2 + (size_t)i * Hc,
                    tokens, tokens, MTOT, Hc, FFc, 0, stream);
    }

    // classifier
    launch_gemm(tokens, w_cls, b_cls, nullptr, clsb, MTOT, NCc, Hc, 0, stream);

    const size_t totOut = (size_t)Bc * NCc + (size_t)Bc * Tc * NCc + (size_t)Bc * TTOT * Hc;
    assemble_out<<<dim3((totOut + 255) / 256), blk, 0, stream>>>(clsb, tokens, outp);
}
